// RNN_att_noise_22651657519486
// MI455X (gfx1250) — compile-verified, matrix operand order checked by reading, not run on hardware
//
#include <hip/hip_runtime.h>
#include <hip/hip_fp16.h>
#include <math.h>

typedef __attribute__((ext_vector_type(16))) _Float16 v16h;
typedef __attribute__((ext_vector_type(8)))  _Float16 v8h;
typedef __attribute__((ext_vector_type(8)))  float    v8f;
typedef __attribute__((ext_vector_type(4)))  float    v4f;

#define Bsz   128
#define Hsz   512
#define Tsz   1024
#define INsz  8
#define NWG   8           // persistent workgroups (one WGP each)
#define NCOL  64          // hidden columns owned per WG
#define BH    (Bsz * Hsz) // 65536

// ---- dynamic LDS layout (bytes) ----
#define OFF_BSLAB  0        // 64 KB : W_rec^T slice in WMMA-B fragment order (f16)
#define OFF_HSLICE 65536    // 32 KB : h slice f32 staging (128 x 64)
#define OFF_XBUF   98304    //  4 KB : x_t  (128 x 8 f32)
#define OFF_WKEY   102400   //  4 KB : W_key (2 x 512 f32)
#define OFF_WIN    106496   //  2 KB : W_in slice (64 x 8 f32)
#define OFF_BIAS   108544   // 256 B : b_in + b_rec slice
#define OFF_BKEY   108800   //   8 B
#define SMEM_BYTES 108864

__device__ __forceinline__ float lik(float tm, float a) {
    float r = (tm - a) / (0.15f * a);
    return (tm == 0.0f) ? 0.0f : r * r;
}

__device__ __forceinline__ float key_dot(const _Float16* __restrict__ hp,
                                         const float* __restrict__ wk, float s) {
    for (int k = 0; k < Hsz; k += 8) {
        v8h hv = *(const v8h*)(hp + k);
#pragma unroll
        for (int u = 0; u < 8; ++u) s = fmaf((float)hv[u], wk[k + u], s);
    }
    return s;
}

__global__ void __launch_bounds__(512, 1)
rnn_scan(const float* __restrict__ inp,  const float* __restrict__ Win,
         const float* __restrict__ bin,  const float* __restrict__ Wrec,
         const float* __restrict__ brec, const float* __restrict__ Wkey,
         const float* __restrict__ bkey, float* __restrict__ out,
         _Float16* __restrict__ hping,   float* __restrict__ stage,
         int* __restrict__ bar,          int use_stage)
{
    extern __shared__ char smem[];
    _Float16* bslab  = (_Float16*)(smem + OFF_BSLAB);
    float*    hslice = (float*)(smem + OFF_HSLICE);
    float*    xbuf   = (float*)(smem + OFF_XBUF);
    float*    wkeyS  = (float*)(smem + OFF_WKEY);
    float*    winS   = (float*)(smem + OFF_WIN);
    float*    biasS  = (float*)(smem + OFF_BIAS);
    float*    bkeyS  = (float*)(smem + OFF_BKEY);

    const int tid   = threadIdx.x;
    const int wg    = blockIdx.x;      // 0..7, owns columns [wg*64, wg*64+64)
    const int lane  = tid & 31;
    const int wave  = tid >> 5;        // 0..15
    const int hi    = lane >> 4;       // lane half (WMMA K / M-row split)
    const int l15   = lane & 15;
    const int mtile = wave >> 1;       // 8 M-tiles of 16 batch rows
    const int npair = wave & 1;        // 2 local N-tile pairs
    const int jl0   = npair * 32 + l15;

    float* outKeys = out;                      // (B,2,T)
    float* outPrs  = out + Bsz * 2 * Tsz;      // (B,4,T)
    float* outHs   = out + Bsz * 6 * Tsz;      // (B,H,T)

    // ---------------- init: stage constant weights into LDS ----------------
    winS[tid] = Win[wg * 512 + tid];                       // 64x8 rows, contiguous
    if (tid < 64) biasS[tid] = bin[wg * 64 + tid] + brec[wg * 64 + tid];
    for (int i = tid; i < 2 * Hsz; i += 512) wkeyS[i] = Wkey[i];
    if (tid < 2) bkeyS[tid] = bkey[tid];

    // Pre-swizzle W_rec^T slice into exact 16x16x32 f16 B-fragment order:
    // frag idx = ((ntile*16 + kstep)*32 + lane)*16 + e ;
    // N = wg*64 + ntile*16 + (lane&15),  K = kstep*32 + (e<8 ? hi*8+e : 16+hi*8+e-8)
    for (int idx = tid; idx < NCOL * Hsz; idx += 512) {
        int e  = idx & 15;
        int l  = (idx >> 4) & 31;
        int ks = (idx >> 9) & 15;
        int nt = idx >> 13;
        int n  = wg * NCOL + nt * 16 + (l & 15);
        int lh = l >> 4;
        int k  = ks * 32 + ((e < 8) ? (lh * 8 + e) : (16 + lh * 8 + (e - 8)));
        bslab[idx] = (_Float16)Wrec[n * Hsz + k];
    }
    __syncthreads();

    // ---------------- sequential scan over t = 1..1023 ----------------
    for (int t = 1; t < Tsz; ++t) {
        // grid barrier: h_{t-1} (f16 ping buffer) visible device-wide
        __threadfence();
        __syncthreads();
        if (tid == 0) {
            __hip_atomic_fetch_add(&bar[t], 1, __ATOMIC_ACQ_REL, __HIP_MEMORY_SCOPE_AGENT);
            while (__hip_atomic_load(&bar[t], __ATOMIC_ACQUIRE, __HIP_MEMORY_SCOPE_AGENT) < NWG)
                __builtin_amdgcn_s_sleep(1);
        }
        __syncthreads();

        const _Float16* hprev = hping + (((t - 1) & 1) * BH);
        _Float16*       hcur  = hping + ((t & 1) * BH);

        // stage x_t into LDS (+ prefetch the strided column a few steps ahead)
        for (int idx = tid; idx < Bsz * INsz; idx += 512) {
            const float* p = inp + (size_t)idx * Tsz + t;
            xbuf[idx] = *p;
            if (t + 8 < Tsz) __builtin_prefetch(p + 8, 0, 0);   // global_prefetch_b8
        }
        __syncthreads();

        // wave0: keys for col (t-1) from h_{t-1}, and likelihoods for col t
        if (wave == 0) {
            int bl = lane >> 1, o = lane & 1;
            int b  = wg * 16 + bl;
            float kv = 1.0f;                                   // key_0 carry = 1
            if (t > 1) {
                float s = key_dot(hprev + b * Hsz, wkeyS + o * Hsz, bkeyS[o]);
                kv = 1.0f / (1.0f + __expf(-s));
                outKeys[(b * 2 + o) * Tsz + (t - 1)] = kv;
            }
            if (o == 0) {                                      // pr_t uses key_{t-1}[b,0]
                const float* x = xbuf + b * INsz;
                float u0 = x[2] * kv + x[4] * (1.0f - kv);
                float u1 = x[3] * kv + x[5] * (1.0f - kv);
                outPrs[(b * 4 + 0) * Tsz + t] = lik(x[6], x[0]);
                outPrs[(b * 4 + 1) * Tsz + t] = lik(x[6], x[1]);
                outPrs[(b * 4 + 2) * Tsz + t] = lik(x[7], u0);
                outPrs[(b * 4 + 3) * Tsz + t] = lik(x[7], u1);
            }
        }

        // ----- GEMM: C = x_t @ W_in^T + bias  (+ h_{t-1} @ W_rec^T via WMMA) -----
        v8f c0 = {}, c1 = {};
#pragma unroll
        for (int r = 0; r < 8; ++r) {                          // C layout: M = r + 8*hi
            int b = mtile * 16 + r + 8 * hi;
            const float* x = xbuf + b * INsz;
            float a0 = biasS[jl0], a1 = biasS[jl0 + 16];
#pragma unroll
            for (int i = 0; i < INsz; ++i) {
                a0 = fmaf(x[i], winS[jl0 * 8 + i], a0);
                a1 = fmaf(x[i], winS[(jl0 + 16) * 8 + i], a1);
            }
            c0[r] = a0; c1[r] = a1;
        }
        if (t > 1) {                                           // h_0 == 0: skip WMMA at t==1
            const _Float16* ha = hprev + (mtile * 16 + l15) * Hsz + hi * 8;
            const int bb0 = ((npair * 2 + 0) * 16) * 512 + lane * 16;
            const int bb1 = ((npair * 2 + 1) * 16) * 512 + lane * 16;
            for (int ks = 0; ks < 16; ++ks) {
                v8h lo = *(const v8h*)(ha + ks * 32);
                v8h hv = *(const v8h*)(ha + ks * 32 + 16);
                v16h a;
#pragma unroll
                for (int e = 0; e < 8; ++e) { a[e] = lo[e]; a[e + 8] = hv[e]; }
                v16h bf0 = *(const v16h*)(bslab + bb0 + ks * 512);
                v16h bf1 = *(const v16h*)(bslab + bb1 + ks * 512);
                c0 = __builtin_amdgcn_wmma_f32_16x16x32_f16(false, a, false, bf0,
                                                            (short)0, c0, false, false);
                c1 = __builtin_amdgcn_wmma_f32_16x16x32_f16(false, a, false, bf1,
                                                            (short)0, c1, false, false);
            }
        }

        // tanh + stage slice in LDS
#pragma unroll
        for (int r = 0; r < 8; ++r) {
            int b = mtile * 16 + r + 8 * hi;
            hslice[b * NCOL + jl0]      = tanhf(c0[r]);
            hslice[b * NCOL + jl0 + 16] = tanhf(c1[r]);
        }
        __syncthreads();

        // coalesced writeback: f16 ping buffer + f32 hs (time-major stage or direct)
        {
            int b = tid >> 2, q = tid & 3;
            const float* src = hslice + b * NCOL + q * 16;
            _Float16* hd = hcur + b * Hsz + wg * NCOL + q * 16;
            v8h h0, h1;
#pragma unroll
            for (int e = 0; e < 8; ++e) { h0[e] = (_Float16)src[e]; h1[e] = (_Float16)src[e + 8]; }
            *(v8h*)(hd)     = h0;
            *(v8h*)(hd + 8) = h1;
            if (use_stage) {
                float* dst = stage + (size_t)t * BH + b * Hsz + wg * NCOL + q * 16;
#pragma unroll
                for (int e = 0; e < 16; e += 4) *(v4f*)(dst + e) = *(const v4f*)(src + e);
            } else {
#pragma unroll
                for (int e = 0; e < 16; ++e) {
                    int jg = wg * NCOL + q * 16 + e;
                    outHs[((size_t)(b * Hsz + jg)) * Tsz + t] = src[e];
                }
            }
        }
    }

    // ---------------- epilogue: keys for final column T-1 ----------------
    __threadfence();
    __syncthreads();
    if (tid == 0) {
        __hip_atomic_fetch_add(&bar[0], 1, __ATOMIC_ACQ_REL, __HIP_MEMORY_SCOPE_AGENT);
        while (__hip_atomic_load(&bar[0], __ATOMIC_ACQUIRE, __HIP_MEMORY_SCOPE_AGENT) < NWG)
            __builtin_amdgcn_s_sleep(1);
    }
    __syncthreads();
    if (wave == 0) {
        int bl = lane >> 1, o = lane & 1;
        int b  = wg * 16 + bl;
        const _Float16* hp = hping + (((Tsz - 1) & 1) * BH) + b * Hsz;
        float s = key_dot(hp, wkeyS + o * Hsz, bkeyS[o]);
        outKeys[(b * 2 + o) * Tsz + (Tsz - 1)] = 1.0f / (1.0f + __expf(-s));
    }
}

// (T, B*H) -> (B*H, T) bandwidth-optimal transpose; emits zeros for t==0.
__global__ void __launch_bounds__(256)
hs_transpose(const float* __restrict__ stage, float* __restrict__ outHs)
{
    __shared__ float tile[32][33];
    int t0 = blockIdx.x * 32;
    int p0 = blockIdx.y * 32;
    int tx = threadIdx.x, ty = threadIdx.y;     // 32 x 8
#pragma unroll
    for (int j = 0; j < 32; j += 8) {
        int t = t0 + ty + j;
        tile[ty + j][tx] = (t == 0) ? 0.0f : stage[(size_t)t * BH + p0 + tx];
    }
    __syncthreads();
#pragma unroll
    for (int j = 0; j < 32; j += 8)
        outHs[(size_t)(p0 + ty + j) * Tsz + t0 + tx] = tile[tx][ty + j];
}

extern "C" void kernel_launch(void* const* d_in, const int* in_sizes, int n_in,
                              void* d_out, int out_size, void* d_ws, size_t ws_size,
                              hipStream_t stream) {
    const float* inp  = (const float*)d_in[0];
    const float* Win  = (const float*)d_in[1];
    const float* bin  = (const float*)d_in[2];
    const float* Wrec = (const float*)d_in[3];
    const float* brec = (const float*)d_in[4];
    const float* Wkey = (const float*)d_in[5];
    const float* bkey = (const float*)d_in[6];
    float* out = (float*)d_out;

    char*     ws    = (char*)d_ws;
    int*      bar   = (int*)ws;                                   //   4 KB barrier slots
    _Float16* hping = (_Float16*)(ws + 4096);                     // 256 KB f16 h ping-pong
    float*    stage = (float*)(ws + 4096 + 2 * BH * (int)sizeof(_Float16));
    size_t stage_need = 4096 + (size_t)2 * BH * 2 + (size_t)Tsz * BH * 4;
    int use_stage = (ws_size >= stage_need) ? 1 : 0;

    hipMemsetAsync(d_ws, 0, 4096, stream);                        // reset grid barrier
    hipMemsetAsync(d_out, 0, (size_t)out_size * sizeof(float), stream); // t==0 cols -> 0

    (void)hipFuncSetAttribute((const void*)rnn_scan,
                              hipFuncAttributeMaxDynamicSharedMemorySize, SMEM_BYTES);

    rnn_scan<<<NWG, 512, SMEM_BYTES, stream>>>(inp, Win, bin, Wrec, brec, Wkey, bkey,
                                               out, hping, stage, bar, use_stage);
    if (use_stage)
        hs_transpose<<<dim3(Tsz / 32, BH / 32), dim3(32, 8), 0, stream>>>(
            stage, out + (size_t)Bsz * 6 * Tsz);
}
